// QLSTM_52664888983990
// MI455X (gfx1250) — matrix operand order repaired and checked by reading, not compiled, not run on hardware
//
#include <hip/hip_runtime.h>
#include <cstddef>
#include <cstdint>

// ---------------------------------------------------------------------------
// QLSTM for MI455X (gfx1250, wave32, WMMA).
// - Fuse x@Wx and h@Wh into one K=1280 bf16 WMMA GEMM per timestep.
// - 32 persistent blocks, each owning a 32-wide hidden slice; cell state c
//   lives in VGPRs for all 256 steps; grid sync via split signal/wait barrier.
// - x-region weights (256 VGPRs) explicitly register-resident across the
//   whole scan; h-region weights stream from L2 each step with an explicit
//   depth-3 B pipeline + depth-4 A pipeline (asm barrier blocks LICM from
//   hoisting 1280 VGPRs of weights and spilling, as seen in prior round).
// - x-region GEMM (K<256) does not depend on h, so it executes BEFORE the
//   barrier wait each step, hiding grid-sync latency.
// ---------------------------------------------------------------------------

typedef __attribute__((ext_vector_type(16))) __bf16 bf16x16;
typedef __attribute__((ext_vector_type(8)))  __bf16 bf16x8;
typedef __attribute__((ext_vector_type(8)))  float  f32x8;

#define T_STEPS 256
#define BATCH   64
#define FEAT    256
#define HID     1024
#define KTOT    1280   /* FEAT + HID */
#define NCLS    257
#define NPAD    288    /* NCLS padded to 32 for 2x2 output blocking */
#define NBLK    32     /* HID / 32 columns per block */

// Workspace layout (bytes), all 256B aligned. Total ~53.2 MB.
#define CNT_OFF   0ull
#define XBF_OFF   256ull
#define WT_OFF    (XBF_OFF + 2ull * T_STEPS * BATCH * FEAT)   /* x bf16: 8 MB      */
#define WOT_OFF   (WT_OFF  + 2ull * 4 * HID * KTOT)           /* W^T bf16: 10.5 MB */
#define HS_OFF    (WOT_OFF + 2ull * NPAD * HID)               /* WoT bf16: 0.58 MB */
/* hs: (T+1) slots of [B][HID] bf16 = 33.7 MB; slot 0 == h_{-1} == 0 */

__device__ __forceinline__ f32x8 zero8() {
  f32x8 z;
#pragma unroll
  for (int i = 0; i < 8; ++i) z[i] = 0.f;
  return z;
}

__device__ __forceinline__ bf16x16 cat8(bf16x8 lo, bf16x8 hi) {
  return __builtin_shufflevector(lo, hi, 0,1,2,3,4,5,6,7,8,9,10,11,12,13,14,15);
}

__device__ __forceinline__ bf16x16 loadA16(const __bf16* p) {
  bf16x8 lo = *(const bf16x8*)p;
  bf16x8 hi = *(const bf16x8*)(p + 8);
  return cat8(lo, hi);
}

__device__ __forceinline__ float sigmoidf_(float x) {
  return 1.f / (1.f + __expf(-x));
}

// ---------------------------------------------------------------------------
// Prep 1: x f32 -> bf16
// ---------------------------------------------------------------------------
__global__ void qlstm_prep_x(const float* __restrict__ x,
                             __bf16* __restrict__ xbf, int n) {
  int i = blockIdx.x * blockDim.x + threadIdx.x;
  if (i < n) xbf[i] = (__bf16)x[i];
}

// ---------------------------------------------------------------------------
// Prep 2: Hamilton-expand [Wx;Wh] into transposed bf16 W^T [4][HID][KTOT].
// Row n contiguous in k so each WMMA B-fragment lane reads 32 contiguous bytes.
// ---------------------------------------------------------------------------
__global__ void qlstm_prep_wcat(const float* __restrict__ Wx,  // [4][4][64][256]
                                const float* __restrict__ Wh,  // [4][4][256][256]
                                __bf16* __restrict__ WT) {
  const int  CMP[4][4] = { {0,1,2,3}, {1,0,3,2}, {2,3,0,1}, {3,2,1,0} };
  const float SGN[4][4] = { { 1.f, 1.f, 1.f, 1.f},
                            {-1.f, 1.f,-1.f, 1.f},
                            {-1.f, 1.f, 1.f,-1.f},
                            {-1.f,-1.f, 1.f, 1.f} };
  long long i = (long long)blockIdx.x * blockDim.x + threadIdx.x;
  const long long N = 4ll * HID * KTOT;
  if (i >= N) return;
  int k = (int)(i % KTOT);
  int n = (int)((i / KTOT) % HID);
  int g = (int)(i / ((long long)KTOT * HID));
  int c  = n >> 8;          // col block (HID/4 = 256)
  int ni = n & 255;
  float v;
  if (k < FEAT) {           // input-to-hidden (FEAT/4 = 64)
    int r = k >> 6, ki = k & 63;
    v = SGN[r][c] * Wx[(((long long)g * 4 + CMP[r][c]) * 64 + ki) * 256 + ni];
  } else {                  // hidden-to-hidden (HID/4 = 256)
    int kh = k - FEAT;
    int r = kh >> 8, ki = kh & 255;
    v = SGN[r][c] * Wh[(((long long)g * 4 + CMP[r][c]) * 256 + ki) * 256 + ni];
  }
  WT[i] = (__bf16)v;
}

// ---------------------------------------------------------------------------
// Prep 3: Wo [HID][NCLS] -> WoT bf16 [NPAD][HID] (transposed, zero padded)
// ---------------------------------------------------------------------------
__global__ void qlstm_prep_wot(const float* __restrict__ Wo,
                               __bf16* __restrict__ WoT) {
  int i = blockIdx.x * blockDim.x + threadIdx.x;
  if (i >= NPAD * HID) return;
  int k = i & (HID - 1);
  int n = i >> 10;
  WoT[i] = (n < NCLS) ? (__bf16)Wo[(long long)k * NCLS + n] : (__bf16)0.f;
}

// ---------------------------------------------------------------------------
// Recurrent scan: 32 persistent blocks x 256 threads (8 waves).
// Block b owns hidden columns [32b, 32b+32). Wave w: mtile = w&3 (16 batch
// rows), ntile = w>>2 (16 cols). Per step: x-region GEMM (pre-barrier,
// register-resident weights), barrier wait, h-region GEMM (pipelined L2
// weight stream), elementwise LSTM update, barrier signal.
// ---------------------------------------------------------------------------
__global__ __launch_bounds__(256, 1)
void qlstm_recurrent(const __bf16* __restrict__ xbf,  // [T][B][FEAT]
                     const __bf16* __restrict__ WT,   // [4][HID][KTOT]
                     const float*  __restrict__ bx,   // [4][HID]
                     __bf16* __restrict__ hs,         // [T+1][B][HID]
                     unsigned* __restrict__ cnt) {
  const int lane  = threadIdx.x & 31;
  const int wave  = threadIdx.x >> 5;
  const int mtile = wave & 3;         // 0..3
  const int ntile = wave >> 2;        // 0..1
  const int lhalf = lane >> 4;
  const int l15   = lane & 15;
  const int ncol  = blockIdx.x * 32 + ntile * 16 + l15;
  const int mrow  = mtile * 16 + l15;

  float bxv[4];
#pragma unroll
  for (int g = 0; g < 4; ++g) bxv[g] = bx[g * HID + ncol];

  // Per-gate, per-lane weight row base pointers.
  const __bf16* wbase[4];
#pragma unroll
  for (int g = 0; g < 4; ++g)
    wbase[g] = WT + ((size_t)g * HID + ncol) * KTOT + lhalf * 16;

  // Explicitly preload the x-region weights once: 8 kk x 4 gates x 8 VGPRs
  // = 256 VGPRs, register-resident for the whole 256-step scan.
  bf16x16 wxf[FEAT / 32][4];
#pragma unroll
  for (int kk = 0; kk < FEAT / 32; ++kk)
#pragma unroll
    for (int g = 0; g < 4; ++g)
      wxf[kk][g] = *(const bf16x16*)(wbase[g] + kk * 32);

  // h-region weight pointers; made opaque each step so the ~1 MB of h-weights
  // streams from L2 per step instead of being LICM-hoisted into (spilled)
  // registers.
  const __bf16* whp[4];
#pragma unroll
  for (int g = 0; g < 4; ++g) whp[g] = wbase[g] + FEAT;

  f32x8 cst = zero8();

  for (int t = 0; t < T_STEPS; ++t) {
    asm volatile("" : "+v"(whp[0]), "+v"(whp[1]), "+v"(whp[2]), "+v"(whp[3]));

    f32x8 acc[4];
#pragma unroll
    for (int g = 0; g < 4; ++g) acc[g] = zero8();

    // ---- x region: K = 0..255 (8 iters), weights already in VGPRs.
    //      Independent of h -> runs BEFORE the barrier wait. ----
    {
      const __bf16* abase = xbf + ((size_t)t * BATCH + mrow) * FEAT + lhalf * 16;
      bf16x16 abuf[4];
#pragma unroll
      for (int i = 0; i < 4; ++i) abuf[i] = loadA16(abase + i * 32);
#pragma unroll
      for (int kk = 0; kk < FEAT / 32; ++kk) {
        bf16x16 a = abuf[kk & 3];
        if (kk + 4 < FEAT / 32) abuf[kk & 3] = loadA16(abase + (kk + 4) * 32);
#pragma unroll
        for (int g = 0; g < 4; ++g) {
          acc[g] = __builtin_amdgcn_wmma_f32_16x16x32_bf16(
              false, a, false, wxf[kk][g], (short)0, acc[g], false, false);
        }
      }
    }

    // ---- barrier WAIT: h_{t-1} (hs slot t) must be complete. For t=0 the
    //      slot was zeroed by a stream-ordered memset. ----
    if (t > 0) {
      if (threadIdx.x == 0) {
        const unsigned target = (unsigned)t * (unsigned)NBLK;
        while (__hip_atomic_load(cnt, __ATOMIC_ACQUIRE,
                                 __HIP_MEMORY_SCOPE_AGENT) < target) {
          __builtin_amdgcn_s_sleep(1);
        }
      }
      __syncthreads();
    }

    // ---- h region: K = 256..1279 (32 iters). Depth-4 A pipeline, depth-3 B
    //      pipeline: loads for kk+3 in flight while kk's 4 WMMAs execute. ----
    {
      const __bf16* abase = hs + ((size_t)t * BATCH + mrow) * HID + lhalf * 16;
      bf16x16 abuf[4];
#pragma unroll
      for (int i = 0; i < 4; ++i) abuf[i] = loadA16(abase + i * 32);
      bf16x16 bb[3][4];
#pragma unroll
      for (int s = 0; s < 3; ++s)
#pragma unroll
        for (int g = 0; g < 4; ++g)
          bb[s][g] = *(const bf16x16*)(whp[g] + s * 32);

#pragma unroll
      for (int kk = 0; kk < HID / 32; ++kk) {
        const int cur = kk % 3;
        bf16x16 a  = abuf[kk & 3];
        bf16x16 b0 = bb[cur][0], b1 = bb[cur][1];
        bf16x16 b2 = bb[cur][2], b3 = bb[cur][3];
        if (kk + 4 < HID / 32) abuf[kk & 3] = loadA16(abase + (kk + 4) * 32);
        if (kk + 3 < HID / 32) {
#pragma unroll
          for (int g = 0; g < 4; ++g)
            bb[cur][g] = *(const bf16x16*)(whp[g] + (kk + 3) * 32);
        }
        acc[0] = __builtin_amdgcn_wmma_f32_16x16x32_bf16(
            false, a, false, b0, (short)0, acc[0], false, false);
        acc[1] = __builtin_amdgcn_wmma_f32_16x16x32_bf16(
            false, a, false, b1, (short)0, acc[1], false, false);
        acc[2] = __builtin_amdgcn_wmma_f32_16x16x32_bf16(
            false, a, false, b2, (short)0, acc[2], false, false);
        acc[3] = __builtin_amdgcn_wmma_f32_16x16x32_bf16(
            false, a, false, b3, (short)0, acc[3], false, false);
      }
    }

    // ---- elementwise LSTM update (gates: f, i, o, a). C-layout: lane holds
    //      n = ncol, m = mtile*16 + 8*lhalf + e for accumulator element e. ----
    const int mbase = mtile * 16 + 8 * lhalf;
#pragma unroll
    for (int e = 0; e < 8; ++e) {
      float fv = sigmoidf_(acc[0][e] + bxv[0]);
      float iv = sigmoidf_(acc[1][e] + bxv[1]);
      float ov = sigmoidf_(acc[2][e] + bxv[2]);
      float av = tanhf(acc[3][e] + bxv[3]);
      float cv = iv * av + fv * cst[e];
      cst[e] = cv;
      float hv = ov * tanhf(cv);
      hs[((size_t)(t + 1) * BATCH + (mbase + e)) * HID + ncol] = (__bf16)hv;
    }

    // ---- barrier SIGNAL: stores fenced, one signal per block. ----
    __threadfence();
    __syncthreads();
    if (threadIdx.x == 0) {
      __hip_atomic_fetch_add(cnt, 1u, __ATOMIC_RELEASE, __HIP_MEMORY_SCOPE_AGENT);
    }
  }
}

// ---------------------------------------------------------------------------
// Output projection: out[T*B][NCLS] = hs @ Wo + bo as a WMMA GEMM.
// M = 16384, K = 1024, N padded to 288. 2x2 C tiles per wave (A/B fragment
// reuse halves L2 traffic vs 1 tile per wave).
// ---------------------------------------------------------------------------
__global__ __launch_bounds__(256, 2)
void qlstm_outproj(const __bf16* __restrict__ hs,   // [T+1][B][HID]
                   const __bf16* __restrict__ WoT,  // [NPAD][HID]
                   const float*  __restrict__ bo,   // [NCLS]
                   float* __restrict__ out) {       // [T*B][NCLS]
  const int wid   = blockIdx.x * 8 + (threadIdx.x >> 5);
  const int lane  = threadIdx.x & 31;
  const int lhalf = lane >> 4;
  const int l15   = lane & 15;
  const int mt2 = wid / (NPAD / 32);     // 0..511  (32 rows each)
  const int nt2 = wid % (NPAD / 32);     // 0..8    (32 cols each)

  f32x8 acc[2][2];
#pragma unroll
  for (int mi = 0; mi < 2; ++mi)
#pragma unroll
    for (int ni = 0; ni < 2; ++ni) acc[mi][ni] = zero8();

  // hs flat row for output row r is (r + BATCH) because slot = r/64 + 1.
  const __bf16* abase0 = hs + (size_t)(mt2 * 32 + l15 + BATCH) * HID + lhalf * 16;
  const __bf16* abase1 = abase0 + (size_t)16 * HID;
  const __bf16* bbase0 = WoT + (size_t)(nt2 * 32 + l15) * HID + lhalf * 16;
  const __bf16* bbase1 = bbase0 + (size_t)16 * HID;

  for (int kk = 0; kk < HID / 32; ++kk) {
    bf16x16 a0 = loadA16(abase0 + kk * 32);
    bf16x16 a1 = loadA16(abase1 + kk * 32);
    bf16x16 b0 = *(const bf16x16*)(bbase0 + kk * 32);
    bf16x16 b1 = *(const bf16x16*)(bbase1 + kk * 32);
    acc[0][0] = __builtin_amdgcn_wmma_f32_16x16x32_bf16(
        false, a0, false, b0, (short)0, acc[0][0], false, false);
    acc[0][1] = __builtin_amdgcn_wmma_f32_16x16x32_bf16(
        false, a0, false, b1, (short)0, acc[0][1], false, false);
    acc[1][0] = __builtin_amdgcn_wmma_f32_16x16x32_bf16(
        false, a1, false, b0, (short)0, acc[1][0], false, false);
    acc[1][1] = __builtin_amdgcn_wmma_f32_16x16x32_bf16(
        false, a1, false, b1, (short)0, acc[1][1], false, false);
  }

#pragma unroll
  for (int ni = 0; ni < 2; ++ni) {
    const int n = nt2 * 32 + ni * 16 + l15;
    if (n < NCLS) {
      const float bv = bo[n];
#pragma unroll
      for (int mi = 0; mi < 2; ++mi) {
        const int mbase = mt2 * 32 + mi * 16 + 8 * lhalf;
#pragma unroll
        for (int e = 0; e < 8; ++e) {
          out[(size_t)(mbase + e) * NCLS + n] = acc[mi][ni][e] + bv;
        }
      }
    }
  }
}

// ---------------------------------------------------------------------------
// Host launch (graph-capture safe: only kernel launches + hipMemsetAsync)
// ---------------------------------------------------------------------------
extern "C" void kernel_launch(void* const* d_in, const int* in_sizes, int n_in,
                              void* d_out, int out_size, void* d_ws, size_t ws_size,
                              hipStream_t stream) {
  const float* x  = (const float*)d_in[0];  // [T][B][FEAT]
  const float* Wx = (const float*)d_in[1];  // [4][4][64][256]
  const float* bx = (const float*)d_in[2];  // [4][HID]
  const float* Wh = (const float*)d_in[3];  // [4][4][256][256]
  const float* Wo = (const float*)d_in[4];  // [HID][NCLS]
  const float* bo = (const float*)d_in[5];  // [NCLS]

  char* ws = (char*)d_ws;
  unsigned* cnt = (unsigned*)(ws + CNT_OFF);
  __bf16* xbf   = (__bf16*)(ws + XBF_OFF);
  __bf16* WT    = (__bf16*)(ws + WT_OFF);
  __bf16* WoT   = (__bf16*)(ws + WOT_OFF);
  __bf16* hs    = (__bf16*)(ws + HS_OFF);

  // Reset barrier counter and h_{-1} slot every launch (deterministic replay).
  hipMemsetAsync(ws + CNT_OFF, 0, 256, stream);
  hipMemsetAsync(ws + HS_OFF, 0, (size_t)BATCH * HID * 2, stream);

  const int xn = T_STEPS * BATCH * FEAT;                 // 4,194,304
  qlstm_prep_x<<<(xn + 255) / 256, 256, 0, stream>>>(x, xbf, xn);

  const long long wn = 4ll * HID * KTOT;                 // 5,242,880
  qlstm_prep_wcat<<<(int)((wn + 255) / 256), 256, 0, stream>>>(Wx, Wh, WT);

  const int won = NPAD * HID;                            // 294,912
  qlstm_prep_wot<<<(won + 255) / 256, 256, 0, stream>>>(Wo, WoT);

  qlstm_recurrent<<<NBLK, 256, 0, stream>>>(xbf, WT, bx, hs, cnt);

  const int nwaves = (T_STEPS * BATCH / 32) * (NPAD / 32);  // 512 * 9 = 4608
  qlstm_outproj<<<nwaves / 8, 256, 0, stream>>>(hs, WoT, bo, (float*)d_out);
}
